// SQuaReFuse_7679401525615
// MI455X (gfx1250) — matrix operand order repaired and checked by reading, not compiled, not run on hardware
//
#include <hip/hip_runtime.h>

#define DEV_INLINE __device__ __forceinline__

typedef __attribute__((ext_vector_type(16))) __bf16 v16bf;
typedef __attribute__((ext_vector_type(8)))  float  v8f;

constexpr int Bb = 8, Tt = 256, Nn = 128, Dd = 256, Kk = 8, Qq = 3, FEAT = 48, Hh = 1024;
constexpr int TCH = 4;                       // t-tiles per wave in proj

union V16BF { v16bf v; unsigned short us[16]; unsigned d[8]; uint4 q[2]; };
union V8F   { v8f  v; float e[8]; };

// Pack two f32 -> one dword of two bf16 (lo in [15:0], hi in [31:16]).
// Round-half-up + v_perm_b32 high-half pack: 3 VALU per pair.
DEV_INLINE unsigned pk2(float lo, float hi) {
#if __has_builtin(__builtin_amdgcn_cvt_pk_bf16_f32)
  auto p = __builtin_amdgcn_cvt_pk_bf16_f32(lo, hi);
  unsigned r; __builtin_memcpy(&r, &p, 4); return r;
#else
  union { float f; unsigned u; } a, b;
  a.f = lo; b.f = hi;
  unsigned ua = a.u + 0x8000u;
  unsigned ub = b.u + 0x8000u;
  // result bytes: [0,1] = ua bytes [2,3]; [2,3] = ub bytes [2,3]
  return __builtin_amdgcn_perm(ub, ua, 0x07060302u);
#endif
}

// Raw 16-float load for one lane of a 16x32 A tile (two 8-float runs).
struct F4x4 { float4 a, b, c, d; };
DEV_INLINE F4x4 load_raw16(const float* __restrict__ rowbase, int dbase) {
  const float4* xp = reinterpret_cast<const float4*>(rowbase + dbase);
  F4x4 r; r.a = xp[0]; r.b = xp[1]; r.c = xp[4]; r.d = xp[5]; return r;
}
DEV_INLINE v16bf pack16(const F4x4& f) {
  V16BF A;
  A.d[0] = pk2(f.a.x, f.a.y); A.d[1] = pk2(f.a.z, f.a.w);
  A.d[2] = pk2(f.b.x, f.b.y); A.d[3] = pk2(f.b.z, f.b.w);
  A.d[4] = pk2(f.c.x, f.c.y); A.d[5] = pk2(f.c.z, f.c.w);
  A.d[6] = pk2(f.d.x, f.d.y); A.d[7] = pk2(f.d.z, f.d.w);
  return A.v;
}
DEV_INLINE v16bf load_a16(const float* __restrict__ rowbase, int dbase) {
  F4x4 f = load_raw16(rowbase, dbase);
  return pack16(f);
}

DEV_INLINE v8f wmma_bf16(v16bf a, v16bf b, v8f c) {
  return __builtin_amdgcn_wmma_f32_16x16x32_bf16(false, a, false, b, (short)0, c, false, false);
}

// ---------------------------------------------------------------------------
// Kernel 1: v[b,n,k,t] = mask[b,t,n] * sum_d x[b,t,n,d] * U[d,k]
// Wave = 16(n) x 16(col) tile, TCH consecutive t per wave.
// B-fragments pre-packed to bf16 in LDS in fragment order [step][lane][16]:
// two ds_load_b128 per WMMA step instead of 64 resident VGPRs (no spills).
// ---------------------------------------------------------------------------
__global__ void proj_kernel(const float* __restrict__ x, const float* __restrict__ mask,
                            const float* __restrict__ U, float* __restrict__ v) {
  __shared__ unsigned Bs[8 * 32 * 8];                 // 8 steps * 32 lanes * 8 dwords = 8 KB
  for (int i = threadIdx.x; i < 8 * 32 * 8; i += blockDim.x) {
    int j = i & 7;                // dword within fragment (elements 2j, 2j+1)
    int l = (i >> 3) & 31;        // lane
    int s = i >> 8;               // k-step
    int col = l & 15;
    int kd = s * 32 + (l >> 4) * 8 + 2 * j + (j >> 2) * 8;
    int cidx = min(col, Kk - 1);
    float v0 = U[kd * Kk + cidx];
    float v1 = U[(kd + 1) * Kk + cidx];
    if (col >= Kk) { v0 = 0.0f; v1 = 0.0f; }
    Bs[i] = pk2(v0, v1);
  }
  __syncthreads();

  int wid  = blockIdx.x * (blockDim.x >> 5) + (threadIdx.x >> 5);
  int lane = threadIdx.x & 31;
  int tc    = wid & (Tt / TCH - 1);                   // 64 chunks
  int ntile = (wid >> 6) & 7;                         // N/16 = 8
  int b     = wid >> 9;
  int n0 = ntile * 16;
  int half = lane >> 4, l16 = lane & 15;

  const uint4* bp = reinterpret_cast<const uint4*>(Bs) + lane * 2;  // + s*64 per step

  V8F cc[TCH];
#pragma unroll
  for (int tl = 0; tl < TCH; ++tl) {
    int t = tc * TCH + tl;
    const float* xrow = x + (size_t)((b * Tt + t) * Nn + (n0 + l16)) * Dd;
    if (tl + 1 < TCH) {  // hint-prefetch next t-tile (same lane row, +N*D floats)
      __builtin_prefetch(xrow + (size_t)Nn * Dd, 0, 0);
    }
    V8F c;
#pragma unroll
    for (int r = 0; r < 8; ++r) c.e[r] = 0.0f;

    // software pipeline: loads of step s+1 issued before pack/WMMA of step s
    F4x4 cur = load_raw16(xrow, half * 8);
#pragma unroll
    for (int s = 0; s < 8; ++s) {
      V16BF Bm;
      Bm.q[0] = bp[s * 64 + 0];
      Bm.q[1] = bp[s * 64 + 1];
      if (s < 7) {
        F4x4 nxt = load_raw16(xrow, (s + 1) * 32 + half * 8);
        c.v = wmma_bf16(pack16(cur), Bm.v, c.v);
        cur = nxt;
      } else {
        c.v = wmma_bf16(pack16(cur), Bm.v, c.v);
      }
    }

    // apply mask for this t now (masks need not persist)
    const float4* mp = reinterpret_cast<const float4*>(
        mask + (size_t)(b * Tt + t) * Nn + n0 + half * 8);
    float4 m0 = mp[0], m1 = mp[1];
    float mv[8] = { m0.x, m0.y, m0.z, m0.w, m1.x, m1.y, m1.z, m1.w };
#pragma unroll
    for (int r = 0; r < 8; ++r) cc[tl].e[r] = c.e[r] * mv[r];
  }

  // Store: float4 along t (tc*TCH is 16B aligned in t).
  if (l16 < Kk) {
#pragma unroll
    for (int r = 0; r < 8; ++r) {
      int n = n0 + half * 8 + r;
      float4 o;
      o.x = cc[0].e[r]; o.y = cc[1].e[r]; o.z = cc[2].e[r]; o.w = cc[3].e[r];
      reinterpret_cast<float4*>(
          v + ((size_t)(b * Nn + n) * Kk + l16) * Tt + tc * TCH)[0] = o;
    }
  }
}

// ---------------------------------------------------------------------------
// Kernel 2: per (b,n): Legendre coeffs (pre-sort), bitonic sort 8 series of
// 256 in LDS, soft quantiles, LayerNorm -> fn[b*N+n, 48]
// ---------------------------------------------------------------------------
DEV_INLINE float block_reduce_256(float val, float* red) {
  int tid = threadIdx.x;
  __syncthreads();
  red[tid] = val;
  __syncthreads();
  for (int s = 128; s > 0; s >>= 1) {
    if (tid < s) red[tid] += red[tid + s];
    __syncthreads();
  }
  return red[0];
}

__global__ void feat_kernel(const float* __restrict__ v,
                            const float* __restrict__ ln_g, const float* __restrict__ ln_b,
                            float* __restrict__ fnout) {
  __shared__ float sv[Kk][Tt];                        // 8 KB
  __shared__ float red[256];
  __shared__ float feats[FEAT];
  __shared__ float stats[2];
  int tid = threadIdx.x;
  int bI = blockIdx.x >> 7;
  int nI = blockIdx.x & (Nn - 1);
  size_t base = (size_t)(bI * Nn + nI) * Kk * Tt;
#pragma unroll
  for (int k = 0; k < Kk; ++k) sv[k][tid] = v[base + (size_t)k * Tt + tid];

  // Legendre basis (L2-normalized over T)
  float tt = -1.0f + 2.0f * (float)tid / (float)(Tt - 1);
  float P0 = 1.0f, P1 = tt, P2 = (3.0f * tt * P1 - P0) * 0.5f;
  float s1 = block_reduce_256(P1 * P1, red);
  float s2 = block_reduce_256(P2 * P2, red);
  float B0 = P0 / sqrtf(fmaxf((float)Tt, 1e-6f));
  float B1 = P1 / sqrtf(fmaxf(s1, 1e-6f));
  float B2 = P2 / sqrtf(fmaxf(s2, 1e-6f));

  // Gaussian rank weights, sigma = max(0.5, T/16) = 16
  const float lv[Qq] = {0.1f, 0.5f, 0.9f};
  float w[Qq];
#pragma unroll
  for (int q = 0; q < Qq; ++q) {
    float dq = ((float)tid - lv[q] * (float)(Tt - 1)) * (1.0f / 16.0f);
    w[q] = __expf(-0.5f * dq * dq);
    float sw = block_reduce_256(w[q], red);
    w[q] /= fmaxf(sw, 1e-6f);
  }

  // Legendre trend coefficients (on UNSORTED series)
  for (int k = 0; k < Kk; ++k) {
    float val = sv[k][tid];
    float c0 = block_reduce_256(val * B0, red);
    float c1 = block_reduce_256(val * B1, red);
    float c2 = block_reduce_256(val * B2, red);
    if (tid == 0) { feats[k * 6 + 3] = c0; feats[k * 6 + 4] = c1; feats[k * 6 + 5] = c2; }
  }

  // Bitonic sort: all 8 series lock-step, ascending
  __syncthreads();
  for (int size = 2; size <= Tt; size <<= 1) {
    for (int stride = size >> 1; stride > 0; stride >>= 1) {
      int i = tid, j = i ^ stride;
      if (j > i) {
        bool asc = ((i & size) == 0);
#pragma unroll
        for (int k = 0; k < Kk; ++k) {
          float va = sv[k][i], vb = sv[k][j];
          bool sw = asc ? (va > vb) : (va < vb);
          if (sw) { sv[k][i] = vb; sv[k][j] = va; }
        }
      }
      __syncthreads();
    }
  }

  // Soft quantiles on sorted series
  for (int k = 0; k < Kk; ++k) {
    float sval = sv[k][tid];
#pragma unroll
    for (int q = 0; q < Qq; ++q) {
      float qs = block_reduce_256(sval * w[q], red);
      if (tid == 0) feats[k * 6 + q] = qs;
    }
  }

  // LayerNorm over the 48 features
  __syncthreads();
  if (tid == 0) {
    float mu = 0.0f;
    for (int i = 0; i < FEAT; ++i) mu += feats[i];
    mu *= (1.0f / FEAT);
    float va = 0.0f;
    for (int i = 0; i < FEAT; ++i) { float dfi = feats[i] - mu; va += dfi * dfi; }
    va *= (1.0f / FEAT);
    stats[0] = mu; stats[1] = rsqrtf(va + 1e-5f);
  }
  __syncthreads();
  if (tid < FEAT) {
    float f = (feats[tid] - stats[0]) * stats[1] * ln_g[tid] + ln_b[tid];
    fnout[(size_t)(bI * Nn + nI) * FEAT + tid] = f;
  }
}

// ---------------------------------------------------------------------------
// Kernel 3: h1 = gelu(fn @ W1 + b1)   [1024,48]x[48,1024], K padded to 64
// ---------------------------------------------------------------------------
__global__ void mlp1_kernel(const float* __restrict__ fn, const float* __restrict__ W1,
                            const float* __restrict__ b1, float* __restrict__ h1) {
  int wid  = blockIdx.x * (blockDim.x >> 5) + (threadIdx.x >> 5);
  int lane = threadIdx.x & 31;
  int nt = wid & 63, mt = wid >> 6;
  int half = lane >> 4, l16 = lane & 15;
  int m = mt * 16 + l16;
  int ncol = nt * 16 + l16;
  const float* arow = fn + (size_t)m * FEAT;

  V8F c;
#pragma unroll
  for (int r = 0; r < 8; ++r) c.e[r] = 0.0f;

  // ---- step 0 (k = 0..31): everything valid
  {
    v16bf A = load_a16(arow, half * 8);
    V16BF Bm;
#pragma unroll
    for (int j = 0; j < 8; ++j) {
      int kk = half * 8 + 2 * j + (j >> 2) * 8;
      Bm.d[j] = pk2(W1[(size_t)kk * Hh + ncol], W1[(size_t)(kk + 1) * Hh + ncol]);
    }
    c.v = wmma_bf16(A, Bm.v, c.v);
  }
  // ---- step 1 (k = 32..63): A run1 is zero at compile time; B clamp+select
  {
    const float4* xp = reinterpret_cast<const float4*>(arow + 32 + half * 8);
    float4 f0 = xp[0], f1 = xp[1];                  // k = 32+half*8 .. +7 (< 48)
    V16BF A;
    A.d[0] = pk2(f0.x, f0.y); A.d[1] = pk2(f0.z, f0.w);
    A.d[2] = pk2(f1.x, f1.y); A.d[3] = pk2(f1.z, f1.w);
    A.d[4] = 0u; A.d[5] = 0u; A.d[6] = 0u; A.d[7] = 0u;
    V16BF Bm;
#pragma unroll
    for (int j = 0; j < 8; ++j) {
      int k0 = 32 + half * 8 + 2 * j + (j >> 2) * 8;
      int k1 = k0 + 1;
      float v0 = W1[(size_t)min(k0, FEAT - 1) * Hh + ncol];
      float v1 = W1[(size_t)min(k1, FEAT - 1) * Hh + ncol];
      v0 = (k0 < FEAT) ? v0 : 0.0f;
      v1 = (k1 < FEAT) ? v1 : 0.0f;
      Bm.d[j] = pk2(v0, v1);
    }
    c.v = wmma_bf16(A.v, Bm.v, c.v);
  }

  float bias = b1[ncol];
#pragma unroll
  for (int r = 0; r < 8; ++r) {
    int mr = mt * 16 + half * 8 + r;
    float xg = c.e[r] + bias;
    h1[(size_t)mr * Hh + ncol] = 0.5f * xg * (1.0f + erff(xg * 0.70710678118f));
  }
}

// ---------------------------------------------------------------------------
// Kernel 4: y = h1 @ W2 + b2   [1024,1024]x[1024,256]
// ---------------------------------------------------------------------------
__global__ void mlp2_kernel(const float* __restrict__ h1, const float* __restrict__ W2,
                            const float* __restrict__ b2, float* __restrict__ y) {
  int wid  = blockIdx.x * (blockDim.x >> 5) + (threadIdx.x >> 5);
  int lane = threadIdx.x & 31;
  int nt = wid & 15, mt = wid >> 4;
  int half = lane >> 4, l16 = lane & 15;
  int m = mt * 16 + l16;
  int ncol = nt * 16 + l16;
  const float* arow = h1 + (size_t)m * Hh;

  V8F c;
#pragma unroll
  for (int r = 0; r < 8; ++r) c.e[r] = 0.0f;

  for (int s = 0; s < Hh / 32; ++s) {
    int d0 = s * 32;
    v16bf A = load_a16(arow, d0 + half * 8);
    int dB = d0 + half * 8;
    V16BF Bm;
#pragma unroll
    for (int j = 0; j < 8; ++j)
      Bm.d[j] = pk2(W2[(size_t)(dB + 2 * j + (j >> 2) * 8) * Dd + ncol],
                    W2[(size_t)(dB + 2 * j + (j >> 2) * 8 + 1) * Dd + ncol]);
    c.v = wmma_bf16(A, Bm.v, c.v);
  }

  float bias = b2[ncol];
#pragma unroll
  for (int r = 0; r < 8; ++r) {
    int mr = mt * 16 + half * 8 + r;
    y[(size_t)mr * Dd + ncol] = c.e[r] + bias;
  }
}

// ---------------------------------------------------------------------------
// Kernel 5: out = x + mask*beta*y   (mask blend algebraically simplified)
// ---------------------------------------------------------------------------
__global__ void resid_kernel(const float* __restrict__ x, const float* __restrict__ mask,
                             const float* __restrict__ y, const float* __restrict__ beta_p,
                             float* __restrict__ out) {
  size_t idx = (size_t)blockIdx.x * blockDim.x + threadIdx.x;   // float4 index
  float beta = beta_p[0];
  int d4 = (int)(idx & 63);
  int n  = (int)((idx >> 6) & (Nn - 1));
  int t  = (int)((idx >> 13) & (Tt - 1));
  int b  = (int)(idx >> 21);
  float4 xv = reinterpret_cast<const float4*>(x)[idx];
  float4 yv = reinterpret_cast<const float4*>(y)[(size_t)(b * Nn + n) * 64 + d4];
  float s = mask[(size_t)(b * Tt + t) * Nn + n] * beta;
  float4 o;
  o.x = xv.x + s * yv.x; o.y = xv.y + s * yv.y;
  o.z = xv.z + s * yv.z; o.w = xv.w + s * yv.w;
  reinterpret_cast<float4*>(out)[idx] = o;
}

// ---------------------------------------------------------------------------
extern "C" void kernel_launch(void* const* d_in, const int* in_sizes, int n_in,
                              void* d_out, int out_size, void* d_ws, size_t ws_size,
                              hipStream_t stream) {
  (void)in_sizes; (void)n_in; (void)out_size; (void)ws_size;
  const float* x    = (const float*)d_in[0];
  const float* mask = (const float*)d_in[1];
  const float* U    = (const float*)d_in[2];
  const float* ln_g = (const float*)d_in[3];
  const float* ln_b = (const float*)d_in[4];
  const float* W1   = (const float*)d_in[5];
  const float* b1   = (const float*)d_in[6];
  const float* W2   = (const float*)d_in[7];
  const float* b2   = (const float*)d_in[8];
  const float* beta = (const float*)d_in[9];
  float* out = (float*)d_out;

  float* v_ws  = (float*)d_ws;                       // B*N*K*T   = 2,097,152
  float* fn_ws = v_ws  + (size_t)Bb * Nn * Kk * Tt;  // B*N*48    =    49,152
  float* h1_ws = fn_ws + (size_t)Bb * Nn * FEAT;     // B*N*H     = 1,048,576
  float* y_ws  = h1_ws + (size_t)Bb * Nn * Hh;       // B*N*D     =   262,144

  // 1) projection: B*(N/16)*(T/TCH) = 4096 waves, 8 waves/block
  proj_kernel<<<512, 256, 0, stream>>>(x, mask, U, v_ws);
  // 2) features + LN: one block per (b,n)
  feat_kernel<<<Bb * Nn, 256, 0, stream>>>(v_ws, ln_g, ln_b, fn_ws);
  // 3) MLP layer 1: 64x64 tiles
  mlp1_kernel<<<512, 256, 0, stream>>>(fn_ws, W1, b1, h1_ws);
  // 4) MLP layer 2: 64x16 tiles
  mlp2_kernel<<<128, 256, 0, stream>>>(h1_ws, W2, b2, y_ws);
  // 5) residual broadcast: 16,777,216 float4
  resid_kernel<<<65536, 256, 0, stream>>>(x, mask, y_ws, beta, out);
}
